// AutoEncoder3D_61409442398868
// MI455X (gfx1250) — compile-verified, matrix operand order checked by reading, not run on hardware
//
#include <hip/hip_runtime.h>
#include <hip/hip_bf16.h>

// ---------------------------------------------------------------------------
// AutoEncoder3D + symmetric chamfer loss for MI455X (gfx1250, wave32, WMMA).
//
//   x[2,1024,3] -> enc(3072->512->128->64) -> z[2,64]
//   zi[128,67]=concat(z,grid) -> dec(67->128->512->3072, tanh) -> Y[2,65536,3]
//   loss = sum_n min_m ||y-s||^2 + sum_m min_n ||y-s||^2
//
// Decoder layers 2/3: v_wmma_f32_16x16x32_f16, B pre-transposed to [N,K] so
// both operands load as 2x global_load_b128 per fragment; K-loop is
// double-buffered so loads for step i+1 overlap the WMMAs of step i.
// ---------------------------------------------------------------------------

typedef __attribute__((ext_vector_type(16))) _Float16 v16h;
typedef __attribute__((ext_vector_type(8)))  float    v8f;

union FragU { uint4 u[2]; v16h h; };

// ----- problem constants ----------------------------------------------------
#define BATCH   2
#define D_IN    1024
#define SPACE   3
#define BOTT    64
#define G       64
#define NGEN    (G * D_IN)          // 65536 generated points per batch
#define ROWS    (BATCH * G)         // 128 decoder rows

// ----- workspace layout (bytes, 256-aligned) --------------------------------
#define OFF_Y      0u                         // 2*65536*3*4   = 1572864
#define OFF_WD3H   1572864u                   // 3072*512*2    = 3145728  (transposed)
#define OFF_WD2H   4718592u                   // 512*128*2     = 131072   (transposed)
#define OFF_D1H    4849664u                   // 128*128*2     = 32768
#define OFF_D2H    4882432u                   // 128*512*2     = 131072
#define OFF_H1     5013504u                   // 2*512*4       = 4096
#define OFF_H2     5017600u                   // 2*128*4       = 1024
#define OFF_Z      5018624u                   // 2*64*4        = 512
#define OFF_SMIN   5019136u                   // 2048*4        = 8192
#define OFF_FWD    5027328u                   // 128*4         = 512

// ---------------------------------------------------------------------------
// init: per-target mins = +inf (uint ordering of positive floats), fwd=0
// ---------------------------------------------------------------------------
__global__ void __launch_bounds__(256) init_kernel(unsigned* sminG, float* fwdPart) {
    const int t = threadIdx.x;
    for (int j = t; j < BATCH * D_IN; j += 256) sminG[j] = 0x7F800000u;
    if (t < 128) fwdPart[t] = 0.0f;
}

// ---------------------------------------------------------------------------
// fp32 W[K,N] -> f16 Wt[N,K], 32x32 LDS-tiled (coalesced read AND write).
// Grid: (N/32, K/32), 256 threads (32x8).
// ---------------------------------------------------------------------------
__global__ void __launch_bounds__(256) cvtT_f16_kernel(const float* __restrict__ in,
                                                       _Float16* __restrict__ out,
                                                       int K, int N) {
    __shared__ _Float16 tile[32][33];
    const int nb = blockIdx.x * 32;
    const int kb = blockIdx.y * 32;
    const int tx = threadIdx.x & 31;
    const int ty = threadIdx.x >> 5;
    #pragma unroll
    for (int r = ty; r < 32; r += 8)
        tile[r][tx] = (_Float16)in[(size_t)(kb + r) * N + nb + tx];
    __syncthreads();
    #pragma unroll
    for (int r = ty; r < 32; r += 8)
        out[(size_t)(nb + r) * K + kb + tx] = tile[tx][r];
}

// ---------------------------------------------------------------------------
// naive fp32 dense + ReLU (encoder; tiny: 2-row GEMMs, coalesced over cols)
// ---------------------------------------------------------------------------
__global__ void __launch_bounds__(256) dense_relu_kernel(const float* __restrict__ in,
                                                         const float* __restrict__ W,
                                                         const float* __restrict__ bias,
                                                         float* __restrict__ out,
                                                         int rows, int cols, int K) {
    const int idx = blockIdx.x * 256 + threadIdx.x;
    if (idx >= rows * cols) return;
    const int r = idx / cols, c = idx % cols;
    float acc = bias[c];
    const float* ip = in + (size_t)r * K;
    const float* wp = W + c;
    for (int k = 0; k < K; ++k) acc = fmaf(ip[k], wp[(size_t)k * cols], acc);
    out[idx] = fmaxf(acc, 0.0f);
}

// ---------------------------------------------------------------------------
// decoder layer 1: concat(z[b], grid[g]) (K=67) @ Wd1[67,128] + bd1, ReLU -> f16
// ---------------------------------------------------------------------------
__global__ void __launch_bounds__(256) dec1_kernel(const float* __restrict__ z,
                                                   const float* __restrict__ grid,
                                                   const float* __restrict__ Wd1,
                                                   const float* __restrict__ bd1,
                                                   _Float16* __restrict__ D1h) {
    const int idx = blockIdx.x * 256 + threadIdx.x;     // 128 rows * 128 cols
    if (idx >= ROWS * 128) return;
    const int r = idx >> 7, j = idx & 127;
    const int b = r >> 6, g = r & 63;
    float acc = bd1[j];
    #pragma unroll 8
    for (int e = 0; e < BOTT; ++e)
        acc = fmaf(z[b * BOTT + e], Wd1[(size_t)e * 128 + j], acc);
    #pragma unroll
    for (int e = 0; e < SPACE; ++e)
        acc = fmaf(grid[g * SPACE + e], Wd1[(size_t)(BOTT + e) * 128 + j], acc);
    D1h[idx] = (_Float16)fmaxf(acc, 0.0f);
}

// ---------------------------------------------------------------------------
// fragment loader: one A fragment + four B fragments (all 16B b128 loads)
// ---------------------------------------------------------------------------
template <int Kv>
__device__ __forceinline__ void load_frags(const _Float16* __restrict__ Arow,
                                           const _Float16* __restrict__ B0,
                                           int kt, FragU& a, FragU (&b)[4]) {
    a.u[0] = *reinterpret_cast<const uint4*>(Arow + kt);
    a.u[1] = *reinterpret_cast<const uint4*>(Arow + kt + 16);
    #pragma unroll
    for (int j = 0; j < 4; ++j) {
        b[j].u[0] = *reinterpret_cast<const uint4*>(B0 + (size_t)(16 * j) * Kv + kt);
        b[j].u[1] = *reinterpret_cast<const uint4*>(B0 + (size_t)(16 * j) * Kv + kt + 16);
    }
}

// ---------------------------------------------------------------------------
// WMMA GEMM: C[M,N] = A[M,K](f16) @ B[K,N] via Bt[N,K](f16), + bias.
// One wave computes a 16x64 strip: 4 accumulators, A fragment reused 4x,
// fragment loads double-buffered across K-steps to hide memory latency.
// Fragment layout per CDNA5 ISA 7.12.2:
//   A: lane(0..15)=row, hi-half lanes carry K+8; e -> K = 16*(e>>3)+(e&7)+8*hi
//   B: lane = column (row of Bt), identical K pattern  -> 2x b128 loads each
//   C/D: VGPR r -> M = r + 8*hi, N = lane&15
// MODE 0: ReLU -> f16 out     MODE 1: tanh -> f32 out
// ---------------------------------------------------------------------------
template <int Mv, int Nv, int Kv, int MODE>
__global__ void __launch_bounds__(128) gemm_wmma_kernel(const _Float16* __restrict__ A,
                                                        const _Float16* __restrict__ Bt,
                                                        const float* __restrict__ bias,
                                                        _Float16* __restrict__ outH,
                                                        float* __restrict__ outF) {
    constexpr int NSTRIP = Nv / 64;
    const int wave = threadIdx.x >> 5;
    const int lane = threadIdx.x & 31;
    const int wid  = blockIdx.x * 4 + wave;
    const int mt   = wid / NSTRIP;          // 16-row tile index
    const int ns   = wid % NSTRIP;          // 64-col strip index
    const int hi   = lane >> 4;
    const int ln   = lane & 15;

    v8f c0 = {}, c1 = {}, c2 = {}, c3 = {};
    const _Float16* Arow = A  + (size_t)(mt * 16 + ln) * Kv + hi * 8;
    const _Float16* B0   = Bt + (size_t)(ns * 64 + ln) * Kv + hi * 8;  // sub-tile j at +j*16*Kv

    FragU a[2];
    FragU b[2][4];
    load_frags<Kv>(Arow, B0, 0, a[0], b[0]);        // prologue

    #pragma unroll
    for (int kt = 0; kt < Kv; kt += 32) {
        const int cur = (kt >> 5) & 1;
        const int nxt = cur ^ 1;
        if (kt + 32 < Kv)
            load_frags<Kv>(Arow, B0, kt + 32, a[nxt], b[nxt]);   // prefetch next step
        c0 = __builtin_amdgcn_wmma_f32_16x16x32_f16(false, a[cur].h, false, b[cur][0].h, (short)0, c0, false, false);
        c1 = __builtin_amdgcn_wmma_f32_16x16x32_f16(false, a[cur].h, false, b[cur][1].h, (short)0, c1, false, false);
        c2 = __builtin_amdgcn_wmma_f32_16x16x32_f16(false, a[cur].h, false, b[cur][2].h, (short)0, c2, false, false);
        c3 = __builtin_amdgcn_wmma_f32_16x16x32_f16(false, a[cur].h, false, b[cur][3].h, (short)0, c3, false, false);
    }

    #pragma unroll
    for (int j = 0; j < 4; ++j) {
        const v8f& c = (j == 0) ? c0 : (j == 1) ? c1 : (j == 2) ? c2 : c3;
        const int n  = ns * 64 + j * 16 + ln;
        const float bv = bias[n];
        #pragma unroll
        for (int r = 0; r < 8; ++r) {
            const int m = mt * 16 + r + hi * 8;
            float v = c[r] + bv;
            if constexpr (MODE == 0) {
                outH[(size_t)m * Nv + n] = (_Float16)fmaxf(v, 0.0f);
            } else {
                outF[(size_t)m * Nv + n] = tanhf(v);
            }
        }
    }
}

// ---------------------------------------------------------------------------
// Fused symmetric chamfer: block = 1024 Y points of one batch (4/thread),
// target cloud staged in LDS as float4; reverse mins via staggered LDS
// atomicMin on positive-float-as-uint (exact, deterministic).
// ---------------------------------------------------------------------------
__global__ void __launch_bounds__(256) chamfer_kernel(const float* __restrict__ Y,
                                                      const float* __restrict__ X,
                                                      unsigned* __restrict__ sminG,
                                                      float* __restrict__ fwdPart) {
    __shared__ float4   sp[D_IN];     // 16 KB
    __shared__ unsigned sm[D_IN];     //  4 KB
    __shared__ float    red[256];     //  1 KB

    const int t     = threadIdx.x;
    const int b     = blockIdx.x >> 6;          // 64 chunks per batch
    const int chunk = blockIdx.x & 63;
    const float* Xb = X + (size_t)b * D_IN * SPACE;
    const float* Yb = Y + (size_t)b * NGEN * SPACE;

    for (int j = t; j < D_IN; j += 256) {
        sp[j] = make_float4(Xb[j * 3], Xb[j * 3 + 1], Xb[j * 3 + 2], 0.0f);
        sm[j] = 0x7F800000u;
    }

    float yx[4], yy[4], yz[4], ymin[4];
    #pragma unroll
    for (int q = 0; q < 4; ++q) {
        const int n = chunk * 1024 + q * 256 + t;
        yx[q] = Yb[n * 3]; yy[q] = Yb[n * 3 + 1]; yz[q] = Yb[n * 3 + 2];
        ymin[q] = 3.4e38f;
    }
    __syncthreads();

    for (int it = 0; it < D_IN; ++it) {
        const int j = (t + it) & (D_IN - 1);    // staggered: bank-conflict free
        const float4 s = sp[j];
        float m4 = 3.4e38f;
        #pragma unroll
        for (int q = 0; q < 4; ++q) {
            const float dx = yx[q] - s.x, dy = yy[q] - s.y, dz = yz[q] - s.z;
            const float d2 = fmaf(dx, dx, fmaf(dy, dy, dz * dz));
            ymin[q] = fminf(ymin[q], d2);
            m4      = fminf(m4, d2);
        }
        atomicMin(&sm[j], __float_as_uint(m4)); // ds_min_u32, no return
    }
    __syncthreads();

    for (int j = t; j < D_IN; j += 256)
        atomicMin(&sminG[b * D_IN + j], sm[j]);

    red[t] = ymin[0] + ymin[1] + ymin[2] + ymin[3];
    __syncthreads();
    for (int s = 128; s > 0; s >>= 1) {
        if (t < s) red[t] += red[t + s];
        __syncthreads();
    }
    if (t == 0) fwdPart[blockIdx.x] = red[0];   // fixed slot: deterministic
}

// ---------------------------------------------------------------------------
// finalize: loss = sum(fwd block partials) + sum(per-target mins)
// ---------------------------------------------------------------------------
__global__ void __launch_bounds__(256) finalize_kernel(const unsigned* __restrict__ sminG,
                                                       const float* __restrict__ fwdPart,
                                                       float* __restrict__ out) {
    __shared__ float red[256];
    const int t = threadIdx.x;
    float a = 0.0f;
    for (int j = t; j < BATCH * D_IN; j += 256) a += __uint_as_float(sminG[j]);
    if (t < 128) a += fwdPart[t];
    red[t] = a;
    __syncthreads();
    for (int s = 128; s > 0; s >>= 1) {
        if (t < s) red[t] += red[t + s];
        __syncthreads();
    }
    if (t == 0) out[0] = red[0];
}

// ---------------------------------------------------------------------------
extern "C" void kernel_launch(void* const* d_in, const int* in_sizes, int n_in,
                              void* d_out, int out_size, void* d_ws, size_t ws_size,
                              hipStream_t stream) {
    const float* x    = (const float*)d_in[0];
    const float* grid = (const float*)d_in[1];
    const float* We1  = (const float*)d_in[2];  const float* be1 = (const float*)d_in[3];
    const float* We2  = (const float*)d_in[4];  const float* be2 = (const float*)d_in[5];
    const float* We3  = (const float*)d_in[6];  const float* be3 = (const float*)d_in[7];
    const float* Wd1  = (const float*)d_in[8];  const float* bd1 = (const float*)d_in[9];
    const float* Wd2  = (const float*)d_in[10]; const float* bd2 = (const float*)d_in[11];
    const float* Wd3  = (const float*)d_in[12]; const float* bd3 = (const float*)d_in[13];

    char* ws = (char*)d_ws;
    float*     Y       = (float*)    (ws + OFF_Y);
    _Float16*  Wd3t    = (_Float16*) (ws + OFF_WD3H);   // [3072,512] f16
    _Float16*  Wd2t    = (_Float16*) (ws + OFF_WD2H);   // [512,128]  f16
    _Float16*  D1h     = (_Float16*) (ws + OFF_D1H);
    _Float16*  D2h     = (_Float16*) (ws + OFF_D2H);
    float*     h1      = (float*)    (ws + OFF_H1);
    float*     h2      = (float*)    (ws + OFF_H2);
    float*     zbuf    = (float*)    (ws + OFF_Z);
    unsigned*  sminG   = (unsigned*) (ws + OFF_SMIN);
    float*     fwdPart = (float*)    (ws + OFF_FWD);

    init_kernel<<<1, 256, 0, stream>>>(sminG, fwdPart);

    // f16 conversion + transpose: Wd2[128,512] -> Wd2t[512,128], Wd3[512,3072] -> Wd3t[3072,512]
    cvtT_f16_kernel<<<dim3(512 / 32, 128 / 32), 256, 0, stream>>>(Wd2, Wd2t, 128, 512);
    cvtT_f16_kernel<<<dim3(3072 / 32, 512 / 32), 256, 0, stream>>>(Wd3, Wd3t, 512, 3072);

    // encoder (fp32, tiny)
    dense_relu_kernel<<<4, 256, 0, stream>>>(x,  We1, be1, h1,   BATCH, 512, D_IN * SPACE);
    dense_relu_kernel<<<1, 256, 0, stream>>>(h1, We2, be2, h2,   BATCH, 128, 512);
    dense_relu_kernel<<<1, 256, 0, stream>>>(h2, We3, be3, zbuf, BATCH, BOTT, 128);

    // decoder
    dec1_kernel<<<64, 256, 0, stream>>>(zbuf, grid, Wd1, bd1, D1h);
    gemm_wmma_kernel<ROWS, 512, 128, 0><<< (ROWS/16)*(512/64)/4, 128, 0, stream>>>(
        D1h, Wd2t, bd2, D2h, nullptr);                                  //  16 blocks
    gemm_wmma_kernel<ROWS, 3072, 512, 1><<< (ROWS/16)*(3072/64)/4, 128, 0, stream>>>(
        D2h, Wd3t, bd3, nullptr, Y);                                    //  96 blocks

    // chamfer + reduction
    chamfer_kernel<<<BATCH * (NGEN / 1024), 256, 0, stream>>>(Y, x, sminG, fwdPart);
    finalize_kernel<<<1, 256, 0, stream>>>(sminG, fwdPart, (float*)d_out);
}
